// SemanticsMultiGranularityHeteroGraph_66133906424420
// MI455X (gfx1250) — matrix operand order repaired and checked by reading, not compiled, not run on hardware
//
#include <hip/hip_runtime.h>
#include <hip/hip_bf16.h>

// ---------------------------------------------------------------------------
// MI455X (gfx1250) hetero-GraphSAGE.
//
//  * All matmuls -> V_WMMA_F32_16X16X32_F16 (f16 in, fp32 accum).
//    A (activations) stay fp32 in memory, converted to f16 while staging into
//    LDS (v_cvt_pk_f16_f32). Weights converted once per call to f16, summed
//    where algebra allows (shared-A root terms), and pre-transposed to [N][K]
//    so hot-loop B staging is a single v8h load + ds_store_b128.
//  * Fused multi-source GEMM: D = sum_i A_i @ B_i + sum(bias) computed with
//    VGPR-resident accumulators across all sources -> each output region is
//    written exactly once (no accumulate re-read passes over C).
//  * Each wave computes a 32x32 C tile: 2 A frags x 2 B frags = 4 WMMAs per
//    32-deep k-step. Block tile 128x64, 8 waves, ~16KB LDS, padded strides
//    to avoid 64-bank conflicts.
//  * segment_mean: wave-per-edge global_atomic_add_f32 scatter into fp32 agg
//    buffers (~267MB, mostly L2-resident on 192MB L2), then in-place divide.
//
// Input index map (setup_inputs dict order, params flattened recursively):
//   0 x_conv 1 x_sent 2 x_word
//   3 Wc 4 bc 5 Ws 6 bs 7 Ww 8 bw
//   9 cs.Wl 10 cs.bl 11 cs.Wr   12 ss.Wl 13 ss.bl 14 ss.Wr
//  15 sw.Wl 16 sw.bl 17 sw.Wr   18 ww.Wl 19 ww.bl 20 ww.Wr
//  21 sc.Wl 22 sc.bl 23 sc.Wr   24 ws.Wl 25 ws.bl 26 ws.Wr
//  27 ei_cs 28 ei_ss 29 ei_sw 30 ei_ww 31 ei_sc 32 ei_ws
// ---------------------------------------------------------------------------

typedef _Float16 v8h  __attribute__((ext_vector_type(8)));
typedef _Float16 v16h __attribute__((ext_vector_type(16)));
typedef float    v4f  __attribute__((ext_vector_type(4)));
typedef float    v8f  __attribute__((ext_vector_type(8)));

#define HID 256
#define TILE_M 128
#define TILE_N 64
#define TILE_K 32
#define SA_STRIDE 40   // halves; 80B rows -> conflict-free frag loads, 16B aligned
#define SB_STRIDE 48   // halves; 96B rows -> 16B-aligned v8h frag loads

// --------------------------------------- weight (sum+)convert+transpose
// Bt[N][K] f16 = (W0 + W1? + W2?)[K][N] fp32
__global__ __launch_bounds__(256) void cvt_sumT_kernel(
    const float* __restrict__ W0, const float* __restrict__ W1,
    const float* __restrict__ W2, _Float16* __restrict__ Bt, int K, int N) {
  int i = blockIdx.x * 256 + threadIdx.x;
  if (i < K * N) {
    float v = W0[i];
    if (W1) v += W1[i];
    if (W2) v += W2[i];
    int k = i / N, n = i - k * N;
    Bt[(long)n * K + k] = (_Float16)v;
  }
}

// ------------------------------------------------------------ fused WMMA GEMM
// C[M,N] = sum_{s<nsrc} A_s[M,K](fp32) @ Bt_s[N,K](f16,transposed) + sum(bias)
// K % 32 == 0, N % 64 == 0, M guarded. C written exactly once.
struct FusedArgs {
  const float*    A[4];
  const _Float16* B[4];
  const float*    bias[3];
  int nsrc, M, N, K;
};

__global__ __launch_bounds__(256) void gemm_fused_wmma(FusedArgs args,
                                                       float* __restrict__ C)
{
  alignas(64) __shared__ _Float16 sA[TILE_M * SA_STRIDE];  // [row][k]
  alignas(64) __shared__ _Float16 sB[TILE_N * SB_STRIDE];  // [n][k]

  const int tid  = threadIdx.x;
  const int lane = tid & 31;
  const int w    = tid >> 5;   // wave 0..7
  const int wr   = w & 3;      // 32-row block within tile
  const int wc   = w >> 2;     // 32-col block within tile

  const int bx = blockIdx.x;   // N tile index
  const int by = blockIdx.y;   // M tile index
  const int M = args.M, N = args.N, K = args.K;

  v8f acc00 = {}, acc01 = {}, acc10 = {}, acc11 = {};

  // staging assignments
  const int ar = tid >> 1;            // A: 128 rows, 2 threads/row
  const int ac = (tid & 1) * 16;      // 16 fp32 each
  const int bn = tid >> 2;            // B: 64 n-rows, 4 threads/row
  const int bk = (tid & 3) * 8;       // 8 f16 each
  const long gRow = (long)by * TILE_M + ar;
  const long gCol = (long)bx * TILE_N + bn;
  const bool aInRange = gRow < M;

  for (int s = 0; s < args.nsrc; ++s) {
    const float*    A  = args.A[s];
    const _Float16* Bt = args.B[s];
    for (int k0 = 0; k0 < K; k0 += TILE_K) {
      // --- stage A (fp32 -> f16)
      {
        v8h h0 = {}, h1 = {};
        if (aInRange) {
          const v4f* p = (const v4f*)(A + gRow * (long)K + k0 + ac);
          v4f x0 = p[0], x1 = p[1], x2 = p[2], x3 = p[3];
#pragma unroll
          for (int i = 0; i < 4; ++i) {
            h0[i]     = (_Float16)x0[i];
            h0[i + 4] = (_Float16)x1[i];
            h1[i]     = (_Float16)x2[i];
            h1[i + 4] = (_Float16)x3[i];
          }
        }
        *(v8h*)(sA + ar * SA_STRIDE + ac)     = h0;
        *(v8h*)(sA + ar * SA_STRIDE + ac + 8) = h1;
      }
      // --- stage B ([N][K]: contiguous both sides)
      {
        const v8h bv = *(const v8h*)(Bt + gCol * K + k0 + bk);
        *(v8h*)(sB + bn * SB_STRIDE + bk) = bv;
      }
      __syncthreads();

      // --- fragments per ISA 7.12.2 wave32 layouts
      const int khalf = lane >> 4;
      const int mr0   = wr * 32 + (lane & 15);
      v16h a0, a1;
#pragma unroll
      for (int i = 0; i < 2; ++i) {
        const _Float16* ap = sA + (mr0 + i * 16) * SA_STRIDE;
        v8h alo = *(const v8h*)(ap + khalf * 8);        // K = khalf*8 + 0..7
        v8h ahi = *(const v8h*)(ap + 16 + khalf * 8);   // K = 16+khalf*8+0..7
        v16h& a = i ? a1 : a0;
#pragma unroll
        for (int j = 0; j < 8; ++j) { a[j] = alo[j]; a[j + 8] = ahi[j]; }
      }
      const int nc = wc * 32 + (lane & 15);
      const v16h b0 = __builtin_shufflevector(*(const v8h*)(sB + nc * SB_STRIDE + khalf * 8), *(const v8h*)(sB + nc * SB_STRIDE + 16 + khalf * 8), 0, 1, 2, 3, 4, 5, 6, 7, 8, 9, 10, 11, 12, 13, 14, 15);
      const v16h b1 = __builtin_shufflevector(*(const v8h*)(sB + (nc + 16) * SB_STRIDE + khalf * 8), *(const v8h*)(sB + (nc + 16) * SB_STRIDE + 16 + khalf * 8), 0, 1, 2, 3, 4, 5, 6, 7, 8, 9, 10, 11, 12, 13, 14, 15);

      acc00 = __builtin_amdgcn_wmma_f32_16x16x32_f16(false, a0, false, b0,
                                                     (short)0, acc00, false, false);
      acc01 = __builtin_amdgcn_wmma_f32_16x16x32_f16(false, a0, false, b1,
                                                     (short)0, acc01, false, false);
      acc10 = __builtin_amdgcn_wmma_f32_16x16x32_f16(false, a1, false, b0,
                                                     (short)0, acc10, false, false);
      acc11 = __builtin_amdgcn_wmma_f32_16x16x32_f16(false, a1, false, b1,
                                                     (short)0, acc11, false, false);
      __syncthreads();
    }
  }

  // --- epilogue: C layout VGPR g -> M = (lane/16)*8 + g, N = lane&15
  const int n  = lane & 15;
  const int mg = (lane >> 4) * 8;
#pragma unroll
  for (int i = 0; i < 2; ++i) {
#pragma unroll
    for (int j = 0; j < 2; ++j) {
      const v8f acc = i ? (j ? acc11 : acc10) : (j ? acc01 : acc00);
      const int col = bx * TILE_N + wc * 32 + j * 16 + n;
      float bv = 0.0f;
#pragma unroll
      for (int b = 0; b < 3; ++b)
        if (args.bias[b]) bv += args.bias[b][col];
#pragma unroll
      for (int g = 0; g < 8; ++g) {
        const int row = by * TILE_M + wr * 32 + i * 16 + mg + g;
        if (row < M) C[(long)row * N + col] = acc[g] + bv;
      }
    }
  }
}

// ------------------------------------------------------------- scatter add
// one wave per edge (8 edges / 256-thread block); lane covers 8 features
__global__ __launch_bounds__(256) void scatter_add_kernel(
    const float* __restrict__ xsrc, const int* __restrict__ src,
    const int* __restrict__ dst, float* __restrict__ agg,
    float* __restrict__ cnt, int E)
{
  const int lane = threadIdx.x & 31;
  const int e = blockIdx.x * 8 + (threadIdx.x >> 5);
  if (e >= E) return;
  const int s = src[e];
  const int d = dst[e];
  const float* xp = xsrc + (long)s * HID + lane * 8;
  float*       ap = agg  + (long)d * HID + lane * 8;
  const v4f x0 = *(const v4f*)xp;
  const v4f x1 = *(const v4f*)(xp + 4);
#pragma unroll
  for (int i = 0; i < 4; ++i) {
    atomicAdd(ap + i,     x0[i]);
    atomicAdd(ap + 4 + i, x1[i]);
  }
  if (lane == 0) atomicAdd(cnt + d, 1.0f);
}

// ----------------------------------------------------------------- mean
__global__ __launch_bounds__(256) void mean_kernel(
    float* __restrict__ agg, const float* __restrict__ cnt, long n)
{
  long i = (long)blockIdx.x * 256 + threadIdx.x;
  if (i < n) {
    float c = cnt[i >> 8];
    agg[i] = agg[i] / fmaxf(c, 1.0f);
  }
}

// ---------------------------------------------------------------------------
extern "C" void kernel_launch(void* const* d_in, const int* in_sizes, int n_in,
                              void* d_out, int out_size, void* d_ws, size_t ws_size,
                              hipStream_t stream) {
  (void)in_sizes; (void)n_in; (void)out_size; (void)ws_size;

  const int N_CONV = 500, N_SENT = 20000, N_WORD = 100000;
  const int E_CS = 20000, E_SS = 160000, E_SW = 200000,
            E_WW = 300000, E_SC = 20000, E_WS = 200000;

  char* base = (char*)d_ws;
  size_t off = 0;
  auto take = [&](size_t elems, size_t esz) -> size_t {
    size_t o = (off + 255) & ~(size_t)255;
    off = o + elems * esz;
    return o;
  };
  auto h16 = [&](size_t elems) { return (_Float16*)(base + take(elems, 2)); };
  auto f32 = [&](size_t elems) { return (float*)(base + take(elems, 4)); };

  // f16 transposed weight buffers [N][K]
  _Float16* hWc     = h16(512 * 256);
  _Float16* hWs     = h16(512 * 256);
  _Float16* hWw     = h16(768 * 256);
  _Float16* hWl_cs  = h16(256 * 256);
  _Float16* hWl_ss  = h16(256 * 256);
  _Float16* hWl_ws  = h16(256 * 256);
  _Float16* hWl_sw  = h16(256 * 256);
  _Float16* hWl_ww  = h16(256 * 256);
  _Float16* hWl_sc  = h16(256 * 256);
  _Float16* hWr_sc  = h16(256 * 256);   // sc.Wr
  _Float16* hWrSumS = h16(256 * 256);   // cs.Wr + ss.Wr + ws.Wr
  _Float16* hWrSumW = h16(256 * 256);   // sw.Wr + ww.Wr

  // projected fp32 features
  float* f_c = f32((size_t)N_CONV * HID);
  float* f_s = f32((size_t)N_SENT * HID);
  float* f_w = f32((size_t)N_WORD * HID);

  // aggregation region (zeroed each call with one memset)
  size_t agg_begin = (off + 255) & ~(size_t)255;
  float* agg_sc = f32((size_t)N_CONV * HID);
  float* agg_cs = f32((size_t)N_SENT * HID);
  float* agg_ss = f32((size_t)N_SENT * HID);
  float* agg_ws = f32((size_t)N_SENT * HID);
  float* agg_sw = f32((size_t)N_WORD * HID);
  float* agg_ww = f32((size_t)N_WORD * HID);
  float* cnt_sc = f32(N_CONV);
  float* cnt_cs = f32(N_SENT);
  float* cnt_ss = f32(N_SENT);
  float* cnt_ws = f32(N_SENT);
  float* cnt_sw = f32(N_WORD);
  float* cnt_ww = f32(N_WORD);
  size_t agg_bytes = off - agg_begin;

  // inputs
  const float* x_conv = (const float*)d_in[0];
  const float* x_sent = (const float*)d_in[1];
  const float* x_word = (const float*)d_in[2];
  const int* ei_cs = (const int*)d_in[27];
  const int* ei_ss = (const int*)d_in[28];
  const int* ei_sw = (const int*)d_in[29];
  const int* ei_ww = (const int*)d_in[30];
  const int* ei_sc = (const int*)d_in[31];
  const int* ei_ws = (const int*)d_in[32];

  float* out_conv = (float*)d_out;
  float* out_sent = out_conv + (size_t)N_CONV * HID;
  float* out_word = out_sent + (size_t)N_SENT * HID;

  auto cvtT = [&](const void* W0, const void* W1, const void* W2,
                  _Float16* dptr, int K, int N) {
    cvt_sumT_kernel<<<(K * N + 255) / 256, 256, 0, stream>>>(
        (const float*)W0, (const float*)W1, (const float*)W2, dptr, K, N);
  };
  auto gemm = [&](FusedArgs args, float* C) {
    dim3 grid(args.N / TILE_N, (args.M + TILE_M - 1) / TILE_M);
    gemm_fused_wmma<<<grid, 256, 0, stream>>>(args, C);
  };
  auto scat = [&](const float* x, const int* ei, float* agg, float* cnt, int E) {
    scatter_add_kernel<<<(E + 7) / 8, 256, 0, stream>>>(x, ei, ei + E, agg, cnt, E);
  };
  auto mean = [&](float* agg, const float* cnt, long nrows) {
    long n = nrows * HID;
    mean_kernel<<<(unsigned)((n + 255) / 256), 256, 0, stream>>>(agg, cnt, n);
  };

  // 1) weights -> f16 transposed [N][K] (root weights for shared-A terms summed)
  cvtT(d_in[3], nullptr, nullptr, hWc, 512, 256);
  cvtT(d_in[5], nullptr, nullptr, hWs, 512, 256);
  cvtT(d_in[7], nullptr, nullptr, hWw, 768, 256);
  cvtT(d_in[9],  nullptr, nullptr, hWl_cs, 256, 256);
  cvtT(d_in[12], nullptr, nullptr, hWl_ss, 256, 256);
  cvtT(d_in[24], nullptr, nullptr, hWl_ws, 256, 256);
  cvtT(d_in[15], nullptr, nullptr, hWl_sw, 256, 256);
  cvtT(d_in[18], nullptr, nullptr, hWl_ww, 256, 256);
  cvtT(d_in[21], nullptr, nullptr, hWl_sc, 256, 256);
  cvtT(d_in[23], nullptr, nullptr, hWr_sc, 256, 256);
  cvtT(d_in[11], d_in[14], d_in[26], hWrSumS, 256, 256);  // cs.Wr+ss.Wr+ws.Wr
  cvtT(d_in[17], d_in[20], nullptr,  hWrSumW, 256, 256);  // sw.Wr+ww.Wr

  // 2) feature projections (WMMA)
  {
    FusedArgs a = {};
    a.nsrc = 1; a.N = HID;
    a.A[0] = x_conv; a.B[0] = hWc; a.bias[0] = (const float*)d_in[4];
    a.M = N_CONV; a.K = 512; gemm(a, f_c);
    a.A[0] = x_sent; a.B[0] = hWs; a.bias[0] = (const float*)d_in[6];
    a.M = N_SENT; a.K = 512; gemm(a, f_s);
    a.A[0] = x_word; a.B[0] = hWw; a.bias[0] = (const float*)d_in[8];
    a.M = N_WORD; a.K = 768; gemm(a, f_w);
  }

  // 3) zero aggregation buffers + counts
  hipMemsetAsync(base + agg_begin, 0, agg_bytes, stream);

  // 4) scatter-add per edge type (src features -> dst aggregation)
  scat(f_s, ei_sc, agg_sc, cnt_sc, E_SC);   // sent -> conv
  scat(f_c, ei_cs, agg_cs, cnt_cs, E_CS);   // conv -> sent
  scat(f_s, ei_ss, agg_ss, cnt_ss, E_SS);   // sent -> sent
  scat(f_w, ei_ws, agg_ws, cnt_ws, E_WS);   // word -> sent
  scat(f_s, ei_sw, agg_sw, cnt_sw, E_SW);   // sent -> word
  scat(f_w, ei_ww, agg_ww, cnt_ww, E_WW);   // word -> word

  // 5) means (in place)
  mean(agg_sc, cnt_sc, N_CONV);
  mean(agg_cs, cnt_cs, N_SENT);
  mean(agg_ss, cnt_ss, N_SENT);
  mean(agg_ws, cnt_ws, N_SENT);
  mean(agg_sw, cnt_sw, N_WORD);
  mean(agg_ww, cnt_ww, N_WORD);

  // 6) fused SAGE linear layers: one single-pass GEMM per destination type
  {
    FusedArgs a = {};   // out_conv = mean_sc@Wl_sc + c@Wr_sc + bl_sc
    a.nsrc = 2; a.M = N_CONV; a.N = HID; a.K = HID;
    a.A[0] = agg_sc; a.B[0] = hWl_sc;
    a.A[1] = f_c;    a.B[1] = hWr_sc;
    a.bias[0] = (const float*)d_in[22];
    gemm(a, out_conv);
  }
  {
    FusedArgs a = {};   // out_sent = sum of 3 SAGEs (root terms merged)
    a.nsrc = 4; a.M = N_SENT; a.N = HID; a.K = HID;
    a.A[0] = agg_cs; a.B[0] = hWl_cs;
    a.A[1] = agg_ss; a.B[1] = hWl_ss;
    a.A[2] = agg_ws; a.B[2] = hWl_ws;
    a.A[3] = f_s;    a.B[3] = hWrSumS;
    a.bias[0] = (const float*)d_in[10];
    a.bias[1] = (const float*)d_in[13];
    a.bias[2] = (const float*)d_in[25];
    gemm(a, out_sent);
  }
  {
    FusedArgs a = {};   // out_word = sum of 2 SAGEs (root terms merged)
    a.nsrc = 3; a.M = N_WORD; a.N = HID; a.K = HID;
    a.A[0] = agg_sw; a.B[0] = hWl_sw;
    a.A[1] = agg_ww; a.B[1] = hWl_ww;
    a.A[2] = f_w;    a.B[2] = hWrSumW;
    a.bias[0] = (const float*)d_in[16];
    a.bias[1] = (const float*)d_in[19];
    gemm(a, out_word);
  }
}
